// GNN4x4_25417616458285
// MI455X (gfx1250) — compile-verified, matrix operand order checked by reading, not run on hardware
//
#include <hip/hip_runtime.h>
#include <hip/hip_bf16.h>
#include <math.h>

typedef _Float16 h16;
typedef _Float16 v16h __attribute__((ext_vector_type(16)));
typedef _Float16 v8h  __attribute__((ext_vector_type(8)));
typedef float    v8f  __attribute__((ext_vector_type(8)));

#define LRELU_SLOPE 0.2f
#define SOFT_EPS    1e-16f

// ---- monotone float<->uint encoding so atomicMax(u32) implements float max ----
__device__ __forceinline__ unsigned enc_f32(float f) {
    unsigned u = __float_as_uint(f);
    return (u & 0x80000000u) ? ~u : (u | 0x80000000u);
}
__device__ __forceinline__ float dec_f32(unsigned u) {
    unsigned b = (u & 0x80000000u) ? (u ^ 0x80000000u) : ~u;
    return __uint_as_float(b);
}

__device__ __forceinline__ float elu_f(float v) {
    return v > 0.f ? v : (__expf(v) - 1.f);
}

// =====================  WMMA GEMM: C[MxN] = A[MxK] * B[KxN]  =====================
// A: f16 row-major [M x K].  BT: f16 [Ncols x K] (B pre-transposed).  C: f32 [M x Ncols].
// One wave per 16x16 output tile; K stepped by 32 with v_wmma_f32_16x16x32_f16.
__global__ void gemm_f16_wmma(const h16* __restrict__ A,
                              const h16* __restrict__ BT,
                              float* __restrict__ C,
                              int M, int K, int Ncols) {
    const int ntiles = Ncols >> 4;
    const int tile = blockIdx.x;
    const int mt = tile / ntiles;
    const int nt = tile % ntiles;
    const int lane = threadIdx.x;       // 0..31 (wave32)
    const int rc = lane & 15;           // A-row / B-col within tile
    const int kh = lane >> 4;           // K-half select (0 or 1)

    int arow = mt * 16 + rc;
    int acl  = arow < M ? arow : (M - 1);          // clamp for ragged M (loads only)
    const h16* pa = A  + (size_t)acl * K + kh * 8;
    const h16* pb = BT + (size_t)(nt * 16 + rc) * K + kh * 8;

    v8f acc = {};
    for (int kt = 0; kt < K; kt += 32) {
        union { v16h v; v8h h[2]; } a, b;
        a.h[0] = *(const v8h*)(pa + kt);        // K = kt + kh*8 + [0..7]
        a.h[1] = *(const v8h*)(pa + kt + 16);   // K = kt + 16 + kh*8 + [0..7]
        b.h[0] = *(const v8h*)(pb + kt);
        b.h[1] = *(const v8h*)(pb + kt + 16);
        acc = __builtin_amdgcn_wmma_f32_16x16x32_f16(
            /*neg_a=*/false, a.v, /*neg_b=*/false, b.v,
            /*c_mod=*/(short)0, acc, /*reuse_a=*/false, /*reuse_b=*/false);
    }

    // C/D layout: VGPR v -> M = kh*8 + v ; N = lane&15
    const int row0 = mt * 16 + kh * 8;
    const int col  = nt * 16 + rc;
    for (int v = 0; v < 8; ++v) {
        int r = row0 + v;
        if (r < M) C[(size_t)r * Ncols + col] = acc[v];
    }
}

// =====================  helpers / elementwise kernels  =====================
__global__ void f32_to_f16_k(const float* __restrict__ in, h16* __restrict__ out, int n) {
    int i = blockIdx.x * blockDim.x + threadIdx.x;
    if (i < n) out[i] = (h16)in[i];
}

// W[rows x cols] (f32, row-major) -> WT[cols x rows] (f16)
__global__ void transpose_to_f16_k(const float* __restrict__ W, h16* __restrict__ WT,
                                   int rows, int cols) {
    int i = blockIdx.x * blockDim.x + threadIdx.x;
    if (i < rows * cols) {
        int r = i / cols, c = i % cols;
        WT[(size_t)c * rows + r] = (h16)W[i];
    }
}

__global__ void zero_u32_k(unsigned* __restrict__ p, int n) {
    int i = blockIdx.x * blockDim.x + threadIdx.x;
    if (i < n) p[i] = 0u;
}

// per-(node,head) attention logits: as = <h[n,hd,:], a_src[hd,:]>, ad likewise
__global__ void alpha_k(const float* __restrict__ h, const float* __restrict__ a_src,
                        const float* __restrict__ a_dst, float* __restrict__ as,
                        float* __restrict__ ad, int N, int H, int D) {
    int i = blockIdx.x * blockDim.x + threadIdx.x;
    if (i >= N * H) return;
    int n = i / H, hd = i % H;
    const float* hp = h + (size_t)n * H * D + (size_t)hd * D;
    const float* ps = a_src + hd * D;
    const float* pd = a_dst + hd * D;
    float s0 = 0.f, s1 = 0.f;
    for (int d = 0; d < D; ++d) { s0 += hp[d] * ps[d]; s1 += hp[d] * pd[d]; }
    as[i] = s0; ad[i] = s1;
}

__device__ __forceinline__ void edge_nodes(const int* __restrict__ ei, int E, int e,
                                           int& s, int& d) {
    if (e < E) { s = ei[e]; d = ei[E + e]; }
    else       { s = e - E; d = e - E; }      // appended self loops
}

// pass 1: segment max over destinations (encoded-uint atomicMax)
__global__ void edge_max_k(const int* __restrict__ ei, int E, int Etot,
                           const float* __restrict__ as, const float* __restrict__ ad,
                           unsigned* __restrict__ mu, int H) {
    int e = blockIdx.x * blockDim.x + threadIdx.x;
    if (e >= Etot) return;
    int s, d; edge_nodes(ei, E, e, s, d);
    for (int h = 0; h < H; ++h) {
        float v = as[s * H + h] + ad[d * H + h];
        v = v >= 0.f ? v : LRELU_SLOPE * v;
        atomicMax(&mu[d * H + h], enc_f32(v));
    }
}

// pass 2: e' = exp(e - max); store per-edge, atomic segment sum
__global__ void edge_expsum_k(const int* __restrict__ ei, int E, int Etot,
                              const float* __restrict__ as, const float* __restrict__ ad,
                              const unsigned* __restrict__ mu, float* __restrict__ ssum,
                              float* __restrict__ eexp, int H) {
    int e = blockIdx.x * blockDim.x + threadIdx.x;
    if (e >= Etot) return;
    int s, d; edge_nodes(ei, E, e, s, d);
    for (int h = 0; h < H; ++h) {
        float v = as[s * H + h] + ad[d * H + h];
        v = v >= 0.f ? v : LRELU_SLOPE * v;
        float ex = __expf(v - dec_f32(mu[d * H + h]));
        eexp[(size_t)e * H + h] = ex;
        atomicAdd(&ssum[d * H + h], ex);
    }
}

// pass 3: alpha = e' / (sum + eps)   (in place)
__global__ void edge_norm_k(const int* __restrict__ ei, int E, int Etot,
                            const float* __restrict__ ssum, float* __restrict__ eexp, int H) {
    int i = blockIdx.x * blockDim.x + threadIdx.x;
    if (i >= Etot * H) return;
    int e = i / H, h = i % H;
    int s, d; edge_nodes(ei, E, e, s, d); (void)s;
    eexp[i] = eexp[i] / (ssum[d * H + h] + SOFT_EPS);
}

// pass 4: out[dst, f] += alpha[e, f/D] * h[src, f]
__global__ void edge_agg_k(const int* __restrict__ ei, int E, int Etot,
                           const float* __restrict__ alpha, const float* __restrict__ hsrc,
                           float* __restrict__ out, int F, int D, int H) {
    long long i = (long long)blockIdx.x * blockDim.x + threadIdx.x;
    if (i >= (long long)Etot * F) return;
    int e = (int)(i / F), f = (int)(i % F);
    int h = f / D;
    int s, d; edge_nodes(ei, E, e, s, d);
    atomicAdd(&out[(size_t)d * F + f], alpha[(size_t)e * H + h] * hsrc[(size_t)s * F + f]);
}

// bias + ELU, emit f16 (input for next WMMA GEMM)
__global__ void bias_elu_f16_k(const float* __restrict__ in, const float* __restrict__ b,
                               h16* __restrict__ out, int N, int F) {
    int i = blockIdx.x * blockDim.x + threadIdx.x;
    if (i >= N * F) return;
    out[i] = (h16)elu_f(in[i] + b[i % F]);
}

// bias + ELU + global mean-pool accumulation
__global__ void pool_k(const float* __restrict__ in, const float* __restrict__ b,
                       const int* __restrict__ batch, float* __restrict__ pool,
                       float* __restrict__ cnt, int N, int D) {
    int i = blockIdx.x * blockDim.x + threadIdx.x;
    if (i >= N * D) return;
    int n = i / D, d = i % D;
    float v = elu_f(in[i] + b[d]);
    int g = batch[n];
    atomicAdd(&pool[(size_t)g * D + d], v);
    if (d == 0) atomicAdd(&cnt[g], 1.0f);
}

// pooled mean -> logits -> log_softmax (2 classes)
__global__ void final_k(const float* __restrict__ pool, const float* __restrict__ cnt,
                        const float* __restrict__ Wfc, const float* __restrict__ bfc,
                        float* __restrict__ out, int G, int D) {
    int g = blockIdx.x * blockDim.x + threadIdx.x;
    if (g >= G) return;
    float c = fmaxf(cnt[g], 1.0f);
    float l0 = bfc[0], l1 = bfc[1];
    for (int d = 0; d < D; ++d) {
        float p = pool[(size_t)g * D + d] / c;
        l0 += p * Wfc[d * 2 + 0];
        l1 += p * Wfc[d * 2 + 1];
    }
    float m = fmaxf(l0, l1);
    float lse = m + logf(__expf(l0 - m) + __expf(l1 - m));
    out[g * 2 + 0] = l0 - lse;
    out[g * 2 + 1] = l1 - lse;
}

// =====================  host orchestration  =====================
static inline int ceil_div_i(long long a, int b) { return (int)((a + b - 1) / b); }

extern "C" void kernel_launch(void* const* d_in, const int* in_sizes, int n_in,
                              void* d_out, int out_size, void* d_ws, size_t ws_size,
                              hipStream_t stream) {
    const float* x      = (const float*)d_in[0];
    const int*   ei     = (const int*)  d_in[1];
    const int*   batch  = (const int*)  d_in[2];
    const float* W1     = (const float*)d_in[3];
    const float* a_src1 = (const float*)d_in[4];
    const float* a_dst1 = (const float*)d_in[5];
    const float* b1     = (const float*)d_in[6];
    const float* W2     = (const float*)d_in[7];
    const float* a_src2 = (const float*)d_in[8];
    const float* a_dst2 = (const float*)d_in[9];
    const float* b2     = (const float*)d_in[10];
    const float* Wfc    = (const float*)d_in[11];
    const float* bfc    = (const float*)d_in[12];
    float* out = (float*)d_out;

    const int N     = in_sizes[2];              // 50000
    const int DIN   = in_sizes[0] / N;          // 128
    const int HID   = in_sizes[10];             // 32
    const int HEADS = in_sizes[4] / HID;        // 4
    const int F1    = HEADS * HID;              // 128
    const int E     = in_sizes[1] / 2;          // 1.6M
    const int Etot  = E + N;                    // + self loops
    const int G     = out_size / 2;             // 64

    // ---- workspace carve-up (256B aligned) ----
    char* wsb = (char*)d_ws;
    size_t off = 0;
    auto carve = [&](size_t bytes) -> char* {
        char* p = wsb + off;
        off += (bytes + 255) & ~(size_t)255;
        return p;
    };
    size_t xh_elems = (size_t)N * (DIN > F1 ? DIN : F1);
    h16*   xh    = (h16*)  carve(xh_elems * sizeof(h16));       // x(f16), reused as conv2 input
    h16*   W1T   = (h16*)  carve((size_t)F1 * DIN * sizeof(h16));
    h16*   W2T   = (h16*)  carve((size_t)HID * F1 * sizeof(h16));
    float* h1    = (float*)carve((size_t)N * F1 * 4);
    float* as1   = (float*)carve((size_t)N * HEADS * 4);
    float* ad1   = (float*)carve((size_t)N * HEADS * 4);
    float* eexp1 = (float*)carve((size_t)Etot * HEADS * 4);
    float* h2    = (float*)carve((size_t)N * HID * 4);
    float* as2   = (float*)carve((size_t)N * 4);
    float* ad2   = (float*)carve((size_t)N * 4);
    float* eexp2 = (float*)carve((size_t)Etot * 4);
    size_t zoff = off;                                           // zero-init region start
    unsigned* m1u  = (unsigned*)carve((size_t)N * HEADS * 4);
    float*    s1   = (float*)   carve((size_t)N * HEADS * 4);
    float*    out1 = (float*)   carve((size_t)N * F1 * 4);
    unsigned* m2u  = (unsigned*)carve((size_t)N * 4);
    float*    s2   = (float*)   carve((size_t)N * 4);
    float*    out2 = (float*)   carve((size_t)N * HID * 4);
    float*    pool = (float*)   carve((size_t)G * HID * 4);
    float*    cnt  = (float*)   carve((size_t)G * 4);
    int zcount = (int)((off - zoff) / 4);

    const int B = 256;

    // stage f16 operands + zero accumulators
    f32_to_f16_k<<<ceil_div_i((long long)N * DIN, B), B, 0, stream>>>(x, xh, N * DIN);
    transpose_to_f16_k<<<ceil_div_i((long long)DIN * F1, B), B, 0, stream>>>(W1, W1T, DIN, F1);
    transpose_to_f16_k<<<ceil_div_i((long long)F1 * HID, B), B, 0, stream>>>(W2, W2T, F1, HID);
    zero_u32_k<<<ceil_div_i(zcount, B), B, 0, stream>>>((unsigned*)(wsb + zoff), zcount);

    // ---- conv1: h1 = x @ W1   (WMMA) ----
    int mt1 = (N + 15) / 16;
    gemm_f16_wmma<<<mt1 * (F1 / 16), 32, 0, stream>>>(xh, W1T, h1, N, DIN, F1);
    alpha_k<<<ceil_div_i((long long)N * HEADS, B), B, 0, stream>>>(h1, a_src1, a_dst1, as1, ad1, N, HEADS, HID);
    edge_max_k   <<<ceil_div_i(Etot, B), B, 0, stream>>>(ei, E, Etot, as1, ad1, m1u, HEADS);
    edge_expsum_k<<<ceil_div_i(Etot, B), B, 0, stream>>>(ei, E, Etot, as1, ad1, m1u, s1, eexp1, HEADS);
    edge_norm_k  <<<ceil_div_i((long long)Etot * HEADS, B), B, 0, stream>>>(ei, E, Etot, s1, eexp1, HEADS);
    edge_agg_k   <<<ceil_div_i((long long)Etot * F1, B), B, 0, stream>>>(ei, E, Etot, eexp1, h1, out1, F1, HID, HEADS);
    bias_elu_f16_k<<<ceil_div_i((long long)N * F1, B), B, 0, stream>>>(out1, b1, xh, N, F1);

    // ---- conv2: h2 = elu(out1+b1) @ W2   (WMMA) ----
    gemm_f16_wmma<<<mt1 * (HID / 16), 32, 0, stream>>>(xh, W2T, h2, N, F1, HID);
    alpha_k<<<ceil_div_i(N, B), B, 0, stream>>>(h2, a_src2, a_dst2, as2, ad2, N, 1, HID);
    edge_max_k   <<<ceil_div_i(Etot, B), B, 0, stream>>>(ei, E, Etot, as2, ad2, m2u, 1);
    edge_expsum_k<<<ceil_div_i(Etot, B), B, 0, stream>>>(ei, E, Etot, as2, ad2, m2u, s2, eexp2, 1);
    edge_norm_k  <<<ceil_div_i(Etot, B), B, 0, stream>>>(ei, E, Etot, s2, eexp2, 1);
    edge_agg_k   <<<ceil_div_i((long long)Etot * HID, B), B, 0, stream>>>(ei, E, Etot, eexp2, h2, out2, HID, HID, 1);

    // ---- pool + classifier ----
    pool_k<<<ceil_div_i((long long)N * HID, B), B, 0, stream>>>(out2, b2, batch, pool, cnt, N, HID);
    final_k<<<ceil_div_i(G, 64), 64, 0, stream>>>(pool, cnt, Wfc, bfc, out, G, HID);
}
